// MaddnessConv2d_62904091018010
// MI455X (gfx1250) — matrix operand order repaired and checked by reading, not compiled, not run on hardware
//
#include <hip/hip_runtime.h>
#include <hip/hip_bf16.h>

typedef __attribute__((ext_vector_type(16))) _Float16 v16h;
typedef __attribute__((ext_vector_type(8)))  _Float16 v8h;
typedef __attribute__((ext_vector_type(8)))  float    v8f;

#define NCB     16
#define HH      56
#define WW      56
#define CIN     64
#define COUT    64
#define BATCH   16
#define NPIX    (HH*WW)          // 3136
#define NPATCH  (BATCH*NPIX)     // 50176
#define NTILES  (NPATCH/16)      // 3136
#define LDS_ROW 264              // halfs per padded LDS row (256 data + 8 pad)

// Prep: BT[n][k] = (f16) lookup_tables[k/16][k%16][n]  (i.e. lut_flat[k*64+n])
__global__ void bt_prep_kernel(const float* __restrict__ lut,
                               _Float16* __restrict__ bt) {
    int q = blockIdx.x * blockDim.x + threadIdx.x;   // 0..16383
    if (q < COUT * 256) {
        int n = q >> 8;
        int k = q & 255;
        bt[q] = (_Float16)lut[k * COUT + n];
    }
}

__global__ __launch_bounds__(256)
void maddness_conv_wmma_kernel(const float* __restrict__ x,
                               const int*   __restrict__ split_idxs,
                               const float* __restrict__ split_vals,
                               const _Float16* __restrict__ bt,
                               const float* __restrict__ bias,
                               float* __restrict__ out) {
    __shared__ __align__(16) _Float16 sBT[COUT * LDS_ROW];  // 33792 bytes

    const int tid = threadIdx.x;

    // ---- stage BT into LDS with padded rows (conflict-free B loads later)
    {
        int n  = tid >> 2;           // 0..63 (table row = output channel)
        int k0 = (tid & 3) << 6;     // 0,64,128,192 halfs within the row
        const float4* src = (const float4*)(bt + n * 256 + k0);
        float4*       dst = (float4*)(sBT + n * LDS_ROW + k0);
        #pragma unroll
        for (int i = 0; i < 8; ++i) dst[i] = src[i];
    }
    __syncthreads();

    const int lane = tid & 31;
    const int wave = tid >> 5;
    const int tile = blockIdx.x * 8 + wave;          // one wave == one 16-patch tile
    if (tile >= NTILES) return;                      // uniform per wave

    const int r    = lane & 15;                      // patch row within tile
    const int hi   = lane >> 4;                      // lane-half select
    const int kb   = hi << 3;                        // A-fragment K base (0 or 8)
    const int nloc = lane & 15;                      // N within N-tile

    // ---- decision-tree encode: lane computes enc for patch (tile*16+r),
    //      codebooks hi*8 .. hi*8+7
    int e[8];
    {
        int p  = tile * 16 + r;
        int ox = p % WW;
        int t1 = p / WW;
        int oy = t1 % HH;
        int b  = t1 / HH;
        const float* xb = x + (size_t)b * CIN * NPIX;
        #pragma unroll
        for (int j = 0; j < 8; ++j) {
            int cb  = hi * 8 + j;
            int enc = 0;
            #pragma unroll
            for (int t = 0; t < 4; ++t) {
                int si = split_idxs[cb * 4 + t];     // 0..35 within subvec
                int ch = cb * 4 + si / 9;            // SUBVEC=36 = 4 ch x 9 kpos
                int kp = si % 9;
                int y  = oy + kp / 3 - 1;            // pad = 1
                int xx = ox + (kp % 3) - 1;
                float v = 0.0f;
                if ((unsigned)y < HH && (unsigned)xx < WW)
                    v = xb[ch * NPIX + y * WW + xx];
                float thr = split_vals[(cb * 4 + t) * 8 + enc];
                enc = 2 * enc + (v >= thr ? 1 : 0);
            }
            e[j] = enc;
        }
    }

    // ---- one-hot GEMM on the matrix pipe: D(16x64) = A(16x256) x BT^T(256x64)
    v8f acc[4] = {};                                 // 4 N-tiles of 16 channels

    const _Float16 one  = (_Float16)1.0f;
    const _Float16 zero = (_Float16)0.0f;

    #pragma unroll
    for (int s = 0; s < 8; ++s) {                    // K step: 32 = 2 codebooks
        // fetch enc[r][2s] and enc[r][2s+1] from the lane that computed them
        const int srcLane = r + ((2 * s) >= 8 ? 16 : 0);
        int ea = __shfl(e[(2 * s)     & 7], srcLane, 32);  // ds_bpermute
        int eb = __shfl(e[(2 * s + 1) & 7], srcLane, 32);

        // synthesize A fragment: elems 0..7 -> K=kb..kb+7 (codebook 2s),
        //                        elems 8..15 -> K=kb+16..kb+23 (codebook 2s+1)
        v16h a;
        #pragma unroll
        for (int h = 0; h < 8; ++h) {
            a[h]     = (ea == kb + h) ? one : zero;
            a[h + 8] = (eb == kb + h) ? one : zero;
        }

        const int koff = 32 * s + kb;                // B K range per lane half
        #pragma unroll
        for (int nt = 0; nt < 4; ++nt) {
            const _Float16* bp = sBT + (nt * 16 + nloc) * LDS_ROW + koff;
            v8h b0 = *(const v8h*)(bp);
            v8h b1 = *(const v8h*)(bp + 16);
            v16h bfrag = __builtin_shufflevector(
                b0, b1, 0, 1, 2, 3, 4, 5, 6, 7, 8, 9, 10, 11, 12, 13, 14, 15);
            acc[nt] = __builtin_amdgcn_wmma_f32_16x16x32_f16(
                false, a, false, bfrag, (short)0, acc[nt], false, false);
        }
    }

    // ---- bias + scattered store (C/D layout: n = lane%16, m = vgpr + 8*hi)
    #pragma unroll
    for (int nt = 0; nt < 4; ++nt) {
        const int ch  = nt * 16 + nloc;
        const float bv = bias[ch];
        #pragma unroll
        for (int i = 0; i < 8; ++i) {
            int m  = i + hi * 8;
            int p  = tile * 16 + m;
            int ox = p % WW;
            int t1 = p / WW;
            int oy = t1 % HH;
            int b  = t1 / HH;
            out[(((size_t)b * COUT + ch) * HH + oy) * WW + ox] = acc[nt][i] + bv;
        }
    }
}

extern "C" void kernel_launch(void* const* d_in, const int* in_sizes, int n_in,
                              void* d_out, int out_size, void* d_ws, size_t ws_size,
                              hipStream_t stream) {
    const float* x          = (const float*)d_in[0];   // (16,64,56,56) f32
    const int*   split_idxs = (const int*)  d_in[1];   // (16,4) i32
    const float* split_vals = (const float*)d_in[2];   // (16,4,8) f32
    const float* lut        = (const float*)d_in[3];   // (16,16,64) f32
    const float* bias       = (const float*)d_in[4];   // (64,) f32
    float*       out        = (float*)d_out;           // (16,64,56,56) f32
    _Float16*    bt         = (_Float16*)d_ws;         // 64x256 f16 = 32KB

    bt_prep_kernel<<<64, 256, 0, stream>>>(lut, bt);
    maddness_conv_wmma_kernel<<<NTILES / 8, 256, 0, stream>>>(
        x, split_idxs, split_vals, bt, bias, out);
}
